// DPAP2PNet_78932908966108
// MI455X (gfx1250) — matrix operand order repaired and checked by reading, not compiled, not run on hardware
//
#include <hip/hip_runtime.h>
#include <math.h>

// ---------------------------------------------------------------------------
// DPAP2PNet head for MI455X (gfx1250, wave32, WMMA).
// Pipeline: NCHW->NHWC bf16 transpose; bilinear gathers (coalesced over C);
// bf16 WMMA GEMMs (v_wmma_f32_16x16x32_bf16) for the two big MLP hidden
// layers; wave-reduction kernels for the tiny (dout=2/7) output layers.
// ---------------------------------------------------------------------------

typedef __attribute__((ext_vector_type(16))) __bf16   v16bf;
typedef __attribute__((ext_vector_type(8)))  float    v8f;
typedef __attribute__((ext_vector_type(4)))  unsigned v4u;

__device__ __forceinline__ unsigned short f2bf(float f) {
  unsigned u = __float_as_uint(f);
  u += 0x7FFFu + ((u >> 16) & 1u);          // round-to-nearest-even
  return (unsigned short)(u >> 16);
}
__device__ __forceinline__ float bf2f(unsigned short h) {
  return __uint_as_float(((unsigned)h) << 16);
}
__device__ __forceinline__ int clampi(int v, int hi) {
  return v < 0 ? 0 : (v > hi ? hi : v);
}

// ---------------------------------------------------------------------------
// 1) NCHW fp32 -> NHWC bf16 (LDS-tiled transpose, coalesced both sides)
// src: [B][256][HW] f32   dst: [B][HW][256] bf16
// ---------------------------------------------------------------------------
__global__ __launch_bounds__(256) void k_nchw_to_nhwc(
    const float* __restrict__ src, unsigned short* __restrict__ dst, int HW) {
  __shared__ float tile[32][33];
  int b   = blockIdx.z;
  int hw0 = blockIdx.x << 5;
  int c0  = blockIdx.y << 5;
  int tx  = threadIdx.x;   // 0..31
  int ty  = threadIdx.y;   // 0..7
  const float* s = src + ((size_t)b * 256 + c0) * (size_t)HW + hw0;
#pragma unroll
  for (int i = 0; i < 32; i += 8)
    tile[ty + i][tx] = s[(size_t)(ty + i) * HW + tx];   // [c_local][hw_local]
  __syncthreads();
  unsigned short* d = dst + ((size_t)b * HW + hw0) * 256 + c0;
#pragma unroll
  for (int i = 0; i < 32; i += 8)
    d[(size_t)(ty + i) * 256 + tx] = f2bf(tile[tx][ty + i]);
}

// ---------------------------------------------------------------------------
// 2) Weight convert+transpose: w[K][N] f32 -> wT[N][K] bf16 (B operand layout)
// ---------------------------------------------------------------------------
__global__ void k_wT_bf16(const float* __restrict__ w,
                          unsigned short* __restrict__ wT, int K, int N) {
  int idx = blockIdx.x * blockDim.x + threadIdx.x;
  if (idx >= K * N) return;
  int k = idx / N, n = idx - k * N;
  wT[(size_t)n * K + k] = f2bf(w[idx]);
}

// ---------------------------------------------------------------------------
// 3) Bilinear point sampling from NHWC bf16 feature map (zeros padding,
//    align_corners=True math: x = px*(S-1)/(stride*S), stride*S == 1024).
//    One block per (b,n) row; thread = channel. Writes bf16 activation row.
// ---------------------------------------------------------------------------
__global__ __launch_bounds__(256) void k_sample(
    const unsigned short* __restrict__ feat,   // [B][S][S][256] bf16
    const float* __restrict__ dp,              // [8192][2] or nullptr (anchors)
    unsigned short* __restrict__ out, int ldo, int colOff, int S, float scale) {
  int row = blockIdx.x;           // b*4096 + n
  int n   = row & 4095;
  int b   = row >> 12;
  float px, py;
  if (dp) { px = dp[2 * row + 0]; py = dp[2 * row + 1]; }
  else    { px = (float)(((n & 63) << 4) + 8); py = (float)(((n >> 6) << 4) + 8); }
  float x = px * scale, y = py * scale;
  float x0f = floorf(x), y0f = floorf(y);
  float fx = x - x0f, fy = y - y0f;
  int ix0 = (int)x0f, iy0 = (int)y0f, ix1 = ix0 + 1, iy1 = iy0 + 1;
  bool vx0 = (ix0 >= 0) && (ix0 < S), vx1 = (ix1 >= 0) && (ix1 < S);
  bool vy0 = (iy0 >= 0) && (iy0 < S), vy1 = (iy1 >= 0) && (iy1 < S);
  float w00 = (vx0 && vy0) ? (1.f - fx) * (1.f - fy) : 0.f;
  float w10 = (vx1 && vy0) ? fx * (1.f - fy) : 0.f;
  float w01 = (vx0 && vy1) ? (1.f - fx) * fy : 0.f;
  float w11 = (vx1 && vy1) ? fx * fy : 0.f;
  int cx0 = clampi(ix0, S - 1), cx1 = clampi(ix1, S - 1);
  int cy0 = clampi(iy0, S - 1), cy1 = clampi(iy1, S - 1);
  size_t bb = (size_t)b * S;
  const unsigned short* p00 = feat + (((bb + cy0) * S + cx0) << 8);
  const unsigned short* p10 = feat + (((bb + cy0) * S + cx1) << 8);
  const unsigned short* p01 = feat + (((bb + cy1) * S + cx0) << 8);
  const unsigned short* p11 = feat + (((bb + cy1) * S + cx1) << 8);
  int c = threadIdx.x;
  float v = w00 * bf2f(p00[c]) + w10 * bf2f(p10[c]) +
            w01 * bf2f(p01[c]) + w11 * bf2f(p11[c]);
  out[(size_t)row * ldo + colOff + c] = f2bf(v);
}

// ---------------------------------------------------------------------------
// 4) bf16 WMMA GEMM: Out[M][256] = relu(A[M][K] @ B[K][256] + bias)
//    A row-major bf16 (lda); BT = B^T as [256][K] bf16.
//    Block = 8 waves; wave computes 16 rows x 128 cols (8 accumulators).
//    A fragment per ISA 16-bit A layout: lane<16 -> K {0..7,16..23},
//    lane>=16 -> K {8..15,24..31}. B fragment: lane holds col=lane%16,
//    same K split as A -> two 16B chunks from BT row.
// ---------------------------------------------------------------------------
__global__ __launch_bounds__(256) void k_gemm_bf16(
    const unsigned short* __restrict__ A, int lda,
    const unsigned short* __restrict__ BT,
    const float* __restrict__ bias,
    float* __restrict__ Out, int K) {
  int lane = threadIdx.x & 31;
  int wave = threadIdx.x >> 5;
  int row0 = (blockIdx.x * 4 + (wave & 3)) * 16;
  int col0 = (wave >> 2) * 128;
  int rowA = row0 + (lane & 15);
  int kA   = (lane & 16) >> 1;    // 0 or 8
  int nB   = lane & 15;
  int kB   = (lane & 16) >> 1;    // 0 or 8
  const unsigned short* aPtr = A + (size_t)rowA * lda + kA;

  v8f acc[8];
#pragma unroll
  for (int t = 0; t < 8; ++t)
#pragma unroll
    for (int d = 0; d < 8; ++d) acc[t][d] = 0.f;

  for (int k = 0; k < K; k += 32) {
    union { v4u u[2]; v16bf v; } ua;
    const v4u* pa = (const v4u*)(aPtr + k);
    ua.u[0] = pa[0];          // K = k+kA .. +7
    ua.u[1] = pa[2];          // K = k+kA+16 .. +23
#pragma unroll
    for (int nt = 0; nt < 8; ++nt) {
      union { v4u u[2]; v16bf v; } ub;
      const v4u* pb = (const v4u*)(BT + (size_t)(col0 + nt * 16 + nB) * K + k + kB);
      ub.u[0] = pb[0];
      ub.u[1] = pb[2];
      acc[nt] = __builtin_amdgcn_wmma_f32_16x16x32_bf16(
          false, ua.v, false, ub.v, (short)0, acc[nt], false, false);
    }
  }

  // C/D layout: VGPR d -> M = d (lanes 0-15) or d+8 (lanes 16-31), N = lane%16
  int mOff = (lane & 16) >> 1;
  int nloc = lane & 15;
#pragma unroll
  for (int nt = 0; nt < 8; ++nt) {
    int ncol = col0 + nt * 16 + nloc;
    float bv = bias[ncol];
#pragma unroll
    for (int d = 0; d < 8; ++d) {
      float v = acc[nt][d] + bv;
      v = v > 0.f ? v : 0.f;                              // ReLU
      Out[(size_t)(row0 + mOff + d) * 256 + ncol] = v;
    }
  }
}

// ---------------------------------------------------------------------------
// 5) Tiny output layer: out[row][off+j] = H[row]·w2[:,j] + b2[j] (+ residual)
//    One wave per row, shuffle reduction. MODE: 0 none, 1 +anchor, 2 +addend.
// ---------------------------------------------------------------------------
template <int J, int MODE>
__global__ __launch_bounds__(256) void k_head2(
    const float* __restrict__ H, const float* __restrict__ w2,
    const float* __restrict__ b2, const float* __restrict__ addend,
    float* __restrict__ out, int ldo, int off) {
  int lane = threadIdx.x & 31;
  int wave = threadIdx.x >> 5;
  int row  = blockIdx.x * 8 + wave;
  const float* h = H + (size_t)row * 256;
  float acc[J];
#pragma unroll
  for (int j = 0; j < J; ++j) acc[j] = 0.f;
#pragma unroll
  for (int i = 0; i < 8; ++i) {
    int k = lane + i * 32;
    float hv = h[k];
#pragma unroll
    for (int j = 0; j < J; ++j) acc[j] = fmaf(hv, w2[k * J + j], acc[j]);
  }
#pragma unroll
  for (int j = 0; j < J; ++j)
#pragma unroll
    for (int m = 16; m >= 1; m >>= 1) acc[j] += __shfl_xor(acc[j], m, 32);
  if (lane == 0) {
#pragma unroll
    for (int j = 0; j < J; ++j) {
      float v = acc[j] + b2[j];
      if (MODE == 1) {
        int n = row & 4095;
        v += (j == 0) ? (float)(((n & 63) << 4) + 8)
                      : (float)(((n >> 6) << 4) + 8);
      } else if (MODE == 2) {
        v += addend[2 * row + j];
      }
      out[(size_t)row * ldo + off + j] = v;
    }
  }
}

// ---------------------------------------------------------------------------
extern "C" void kernel_launch(void* const* d_in, const int* in_sizes, int n_in,
                              void* d_out, int out_size, void* d_ws, size_t ws_size,
                              hipStream_t stream) {
  (void)in_sizes; (void)n_in; (void)out_size; (void)ws_size;
  const float* feat[4] = {(const float*)d_in[0], (const float*)d_in[1],
                          (const float*)d_in[2], (const float*)d_in[3]};
  const float* w_def1 = (const float*)d_in[4];
  const float* b_def1 = (const float*)d_in[5];
  const float* w_def2 = (const float*)d_in[6];
  const float* b_def2 = (const float*)d_in[7];
  const float* w_reg1 = (const float*)d_in[8];
  const float* b_reg1 = (const float*)d_in[9];
  const float* w_reg2 = (const float*)d_in[10];
  const float* b_reg2 = (const float*)d_in[11];
  const float* w_cls1 = (const float*)d_in[12];
  const float* b_cls1 = (const float*)d_in[13];
  const float* w_cls2 = (const float*)d_in[14];
  const float* b_cls2 = (const float*)d_in[15];

  const int B = 2, ROWS = 8192;
  const int S[4] = {256, 128, 64, 32};

  char* ws = (char*)d_ws;
  size_t off = 0;
  auto take = [&](size_t bytes) -> char* {
    char* p = ws + off;
    off = (off + bytes + 255) & ~(size_t)255;
    return p;
  };
  unsigned short* nhwc[4];
  for (int l = 0; l < 4; ++l)
    nhwc[l] = (unsigned short*)take((size_t)B * S[l] * S[l] * 256 * 2);
  unsigned short* wdefT = (unsigned short*)take((size_t)256 * 256 * 2);
  unsigned short* wregT = (unsigned short*)take((size_t)1024 * 256 * 2);
  unsigned short* wclsT = (unsigned short*)take((size_t)1024 * 256 * 2);
  float*          dp    = (float*)take((size_t)ROWS * 2 * 4);
  unsigned short* flat  = (unsigned short*)take((size_t)ROWS * 1024 * 2);
  unsigned short* roi0  = flat;  // roi0 lifetime ends before flat is written
  float*          Hbuf  = (float*)take((size_t)ROWS * 256 * 4);  // reused 3x

  // 1) NCHW->NHWC bf16 transposes
  for (int l = 0; l < 4; ++l) {
    int HW = S[l] * S[l];
    hipLaunchKernelGGL(k_nchw_to_nhwc, dim3(HW / 32, 8, B), dim3(32, 8, 1), 0,
                       stream, feat[l], nhwc[l], HW);
  }
  // 2) weight transposes to bf16
  hipLaunchKernelGGL(k_wT_bf16, dim3(256), dim3(256), 0, stream, w_def1, wdefT, 256, 256);
  hipLaunchKernelGGL(k_wT_bf16, dim3(1024), dim3(256), 0, stream, w_reg1, wregT, 1024, 256);
  hipLaunchKernelGGL(k_wT_bf16, dim3(1024), dim3(256), 0, stream, w_cls1, wclsT, 1024, 256);
  // 3) roi0 = sample(feat0, anchors); scale = (S-1)/(stride*S) = 255/1024
  hipLaunchKernelGGL(k_sample, dim3(ROWS), dim3(256), 0, stream,
                     nhwc[0], (const float*)nullptr, roi0, 256, 0, 256, 255.f / 1024.f);
  // 4) def hidden: H = relu(roi0 @ w_def1 + b_def1)   [8192,256]x[256,256]
  hipLaunchKernelGGL(k_gemm_bf16, dim3(ROWS / 64), dim3(256), 0, stream,
                     roi0, 256, wdefT, b_def1, Hbuf, 256);
  // 5) dp = anchors + H @ w_def2 + b_def2
  hipLaunchKernelGGL((k_head2<2, 1>), dim3(ROWS / 8), dim3(256), 0, stream,
                     Hbuf, w_def2, b_def2, (const float*)nullptr, dp, 2, 0);
  // 6) flat[:, l*256:(l+1)*256] = sample(feat_l, dp)
  for (int l = 0; l < 4; ++l) {
    float scale = (float)(S[l] - 1) / 1024.f;   // stride*S == 1024 at all levels
    hipLaunchKernelGGL(k_sample, dim3(ROWS), dim3(256), 0, stream,
                       nhwc[l], (const float*)dp, flat, 1024, l * 256, S[l], scale);
  }
  // 7) reg hidden + coords (residual dp) -> out[:, 0:2]
  hipLaunchKernelGGL(k_gemm_bf16, dim3(ROWS / 64), dim3(256), 0, stream,
                     flat, 1024, wregT, b_reg1, Hbuf, 1024);
  hipLaunchKernelGGL((k_head2<2, 2>), dim3(ROWS / 8), dim3(256), 0, stream,
                     Hbuf, w_reg2, b_reg2, dp, (float*)d_out, 9, 0);
  // 8) cls hidden + logits -> out[:, 2:9]
  hipLaunchKernelGGL(k_gemm_bf16, dim3(ROWS / 64), dim3(256), 0, stream,
                     flat, 1024, wclsT, b_cls1, Hbuf, 1024);
  hipLaunchKernelGGL((k_head2<7, 0>), dim3(ROWS / 8), dim3(256), 0, stream,
                     Hbuf, w_cls2, b_cls2, (const float*)nullptr, (float*)d_out, 9, 2);
}
